// ScaledDotProductAttention_80461917323588
// MI455X (gfx1250) — matrix operand order repaired and checked by reading, not compiled, not run on hardware
//
#include <hip/hip_runtime.h>

typedef _Float16     v16h __attribute__((ext_vector_type(16)));
typedef _Float16     v8h  __attribute__((ext_vector_type(8)));
typedef float        v8f  __attribute__((ext_vector_type(8)));
typedef unsigned int v4u  __attribute__((ext_vector_type(4)));
typedef unsigned int v8u  __attribute__((ext_vector_type(8)));

#define BH_TOT 32      // B*H
#define SEQ    2048
#define HD     64
#define QTILE  16
#define KBLK   32
#define WAVES_PER_BLOCK 8
#define LDS_STRIDE 40

#define ELEMS ((size_t)BH_TOT * SEQ * HD)   // 4,194,304 per tensor

// ---------------------------------------------------------------------------
// Prep kernel 1: K f32 -> f16, same [bh][key][d] layout
// ---------------------------------------------------------------------------
__global__ __launch_bounds__(256)
void prep_kh_kernel(const float* __restrict__ K, _Float16* __restrict__ Kh)
{
    const size_t i8 = ((size_t)blockIdx.x * 256 + threadIdx.x) * 8;
    const float4* src = (const float4*)(K + i8);
    const float4 a = src[0];
    const float4 b = src[1];
    v8h o;
    o[0] = (_Float16)a.x; o[1] = (_Float16)a.y; o[2] = (_Float16)a.z; o[3] = (_Float16)a.w;
    o[4] = (_Float16)b.x; o[5] = (_Float16)b.y; o[6] = (_Float16)b.z; o[7] = (_Float16)b.w;
    *(v8h*)(Kh + i8) = o;
}

// ---------------------------------------------------------------------------
// Prep kernel 2: V f32 [bh][key][d] -> f16 transposed Vt [bh][d][key]
// ---------------------------------------------------------------------------
__global__ __launch_bounds__(256)
void prep_vt_kernel(const float* __restrict__ V, _Float16* __restrict__ Vt)
{
    __shared__ _Float16 tile[64][65];
    const int bid = blockIdx.x;          // 32 bh * 32 keyblocks = 1024
    const int bh  = bid >> 5;
    const int kb  = (bid & 31) << 6;     // 64 keys per tile
    const int t   = threadIdx.x;

    const float* vbase = V + ((size_t)bh * SEQ + kb) * HD;
#pragma unroll
    for (int i = 0; i < 4; ++i) {
        const int f   = t + i * 256;
        const int key = f >> 4;
        const int dq  = (f & 15) << 2;
        const float4 x = ((const float4*)vbase)[f];
        tile[key][dq + 0] = (_Float16)x.x;
        tile[key][dq + 1] = (_Float16)x.y;
        tile[key][dq + 2] = (_Float16)x.z;
        tile[key][dq + 3] = (_Float16)x.w;
    }
    __syncthreads();

    const int d   = t >> 2;
    const int seg = (t & 3) << 4;
    v16h val;
#pragma unroll
    for (int j = 0; j < 16; ++j) val[j] = tile[seg + j][d];
    *(v16h*)(Vt + ((size_t)bh * HD + d) * SEQ + kb + seg) = val;
}

// ---------------------------------------------------------------------------
// One 32-key block, transposed formulation: S^T = K*Q^T, O^T = V^T*P^T.
// Softmax row = per-lane (lane = query) + one xor-16 exchange.
// MASK=true only for the final (diagonal) block of each wave.
// ---------------------------------------------------------------------------
template<bool MASK>
__device__ __forceinline__
void fa_block_t(const int kb, const int q0, const int col, const int hi,
                const _Float16* __restrict__ kbh, const _Float16* __restrict__ vtb,
                const v16h (&bQ)[2], v8f (&acc)[4],
                float& rowmax, float& rowsum)
{
    // ---- A-fragments of K (16 keys x 32 d): two aligned 16B runs per lane ----
    v16h aK[2][2];
#pragma unroll
    for (int t = 0; t < 2; ++t) {
#pragma unroll
        for (int c = 0; c < 2; ++c) {
            const _Float16* kp = kbh + (size_t)(kb + t * 16 + col) * HD + c * 32 + hi * 16;
            const v8h lo = *(const v8h*)(kp);
            const v8h hh = *(const v8h*)(kp + 8);
#pragma unroll
            for (int i = 0; i < 8; ++i) { aK[t][c][i] = lo[i]; aK[t][c][i + 8] = hh[i]; }
        }
    }

    // ---- S^T tiles: st[t] C-layout = lane -> query col, VGPR r -> key r+8*hi ----
    v8f st0 = (v8f){}, st1 = (v8f){};
    st0 = __builtin_amdgcn_wmma_f32_16x16x32_f16(false, aK[0][0], false, bQ[0], (short)0, st0, false, false);
    st0 = __builtin_amdgcn_wmma_f32_16x16x32_f16(false, aK[0][1], false, bQ[1], (short)0, st0, false, false);
    st1 = __builtin_amdgcn_wmma_f32_16x16x32_f16(false, aK[1][0], false, bQ[0], (short)0, st1, false, false);
    st1 = __builtin_amdgcn_wmma_f32_16x16x32_f16(false, aK[1][1], false, bQ[1], (short)0, st1, false, false);

    if (MASK) {
        const int qcol = q0 + col;
#pragma unroll
        for (int r = 0; r < 8; ++r) {
            if (kb + 8 * hi + r      > qcol) st0[r] = -3.0e38f;
            if (kb + 16 + 8 * hi + r > qcol) st1[r] = -3.0e38f;
        }
    }

    // ---- online softmax: in-lane trees + single cross-half exchange ----
    float a0 = fmaxf(fmaxf(st0[0], st0[1]), fmaxf(st0[2], st0[3]));
    float a1 = fmaxf(fmaxf(st0[4], st0[5]), fmaxf(st0[6], st0[7]));
    float a2 = fmaxf(fmaxf(st1[0], st1[1]), fmaxf(st1[2], st1[3]));
    float a3 = fmaxf(fmaxf(st1[4], st1[5]), fmaxf(st1[6], st1[7]));
    float mt = fmaxf(fmaxf(a0, a1), fmaxf(a2, a3));
    mt = fmaxf(mt, __shfl_xor(mt, 16, 32));

    const float mnew = fmaxf(rowmax, mt);
    const float corr = __expf(rowmax - mnew);
    rowmax = mnew;

#pragma unroll
    for (int r = 0; r < 8; ++r) {
        st0[r] = __expf(st0[r] - mnew);
        st1[r] = __expf(st1[r] - mnew);
    }

    float s0 = (st0[0] + st0[1]) + (st0[2] + st0[3]);
    float s1 = (st0[4] + st0[5]) + (st0[6] + st0[7]);
    float s2 = (st1[0] + st1[1]) + (st1[2] + st1[3]);
    float s3 = (st1[4] + st1[5]) + (st1[6] + st1[7]);
    float ps = (s0 + s1) + (s2 + s3);
    ps += __shfl_xor(ps, 16, 32);
    rowsum = rowsum * corr + ps;

#pragma unroll
    for (int n = 0; n < 4; ++n) acc[n] *= corr;   // corr is lane-uniform per query

    // ---- build P^T B-fragment: own halves + 4 packed-dword xor-16 shuffles ----
    v8h own0, own1;
#pragma unroll
    for (int r = 0; r < 8; ++r) { own0[r] = (_Float16)st0[r]; own1[r] = (_Float16)st1[r]; }
    const v4u o0 = __builtin_bit_cast(v4u, own0);
    const v4u o1 = __builtin_bit_cast(v4u, own1);
    const v4u send = hi ? o0 : o1;                 // partner needs the other tile
    v4u recv;
#pragma unroll
    for (int i = 0; i < 4; ++i) recv[i] = (unsigned)__shfl_xor((int)send[i], 16, 32);
    const v4u elo = hi ? recv : o0;                // keys {0..7} for lo-lanes, {16..23} for hi
    const v4u ehi = hi ? o1 : recv;                // keys {8..15}->partner / {24..31}
    v8u bp;
#pragma unroll
    for (int i = 0; i < 4; ++i) { bp[i] = elo[i]; bp[i + 4] = ehi[i]; }
    const v16h bP = __builtin_bit_cast(v16h, bp);

    // ---- O^T += V^T * P^T : A-fragment of V^T from Vt [d][key], aligned runs ----
#pragma unroll
    for (int n = 0; n < 4; ++n) {
        const _Float16* vp = vtb + (size_t)(n * 16 + col) * SEQ + kb + hi * 16;
        const v8h lo = *(const v8h*)(vp);
        const v8h hh = *(const v8h*)(vp + 8);
        v16h aV;
#pragma unroll
        for (int i = 0; i < 8; ++i) { aV[i] = lo[i]; aV[i + 8] = hh[i]; }
        acc[n] = __builtin_amdgcn_wmma_f32_16x16x32_f16(false, aV, false, bP, (short)0, acc[n], false, false);
    }
}

// ---------------------------------------------------------------------------
// Main flash-attention kernel (fast path, transposed formulation, no LDS).
// ---------------------------------------------------------------------------
__global__ __launch_bounds__(256)
void fa_fwd_causal_ws_kernel(const float* __restrict__ Q, const _Float16* __restrict__ Kh,
                             const _Float16* __restrict__ Vt, float* __restrict__ O)
{
    const int lane = threadIdx.x & 31;
    const int w    = threadIdx.x >> 5;
    const int wid  = blockIdx.x * WAVES_PER_BLOCK + w;
    const int bh   = wid >> 7;
    const int q0   = (wid & 127) << 4;

    const int col = lane & 15;   // query index within tile (both half-waves)
    const int hi  = lane >> 4;

    const float scale = 0.125f;  // 1/sqrt(64)

    // ---- Q^T B-fragments (pre-scaled): lane -> query col, K-rows -> d ----
    const float* qp = Q + ((size_t)bh * SEQ + q0 + col) * HD + hi * 16;
    v16h bQ[2];
#pragma unroll
    for (int c = 0; c < 2; ++c) {
        const float4* q4 = (const float4*)(qp + c * 32);
#pragma unroll
        for (int v4i = 0; v4i < 4; ++v4i) {
            const float4 f = q4[v4i];
            bQ[c][v4i * 4 + 0] = (_Float16)(f.x * scale);
            bQ[c][v4i * 4 + 1] = (_Float16)(f.y * scale);
            bQ[c][v4i * 4 + 2] = (_Float16)(f.z * scale);
            bQ[c][v4i * 4 + 3] = (_Float16)(f.w * scale);
        }
    }

    v8f acc[4];
#pragma unroll
    for (int n = 0; n < 4; ++n) acc[n] = (v8f){};
    float rowmax = -3.0e38f, rowsum = 0.0f;

    const _Float16* kbh = Kh + (size_t)bh * SEQ * HD;   // [key][d]
    const _Float16* vtb = Vt + (size_t)bh * HD * SEQ;   // [d][key]

    // Only the FINAL 32-key block is masked for a 16-row query tile.
    const int kb_last = ((q0 + QTILE - 1) / KBLK) * KBLK;

    for (int kb = 0; kb < kb_last; kb += KBLK) {
        __builtin_prefetch(kbh + (size_t)(kb + KBLK + col) * HD + hi * 8, 0, 3);
        __builtin_prefetch(vtb + (size_t)col * SEQ + kb + KBLK + hi * 8, 0, 3);
        fa_block_t<false>(kb, q0, col, hi, kbh, vtb, bQ, acc, rowmax, rowsum);
    }
    fa_block_t<true>(kb_last, q0, col, hi, kbh, vtb, bQ, acc, rowmax, rowsum);

    // ---- normalize and store: O^T C-layout -> contiguous float4 runs ----
    const float inv = 1.0f / rowsum;
    float* orow = O + ((size_t)bh * SEQ + q0 + col) * HD + hi * 8;
#pragma unroll
    for (int n = 0; n < 4; ++n) {
        float4 lo4, hi4;
        lo4.x = acc[n][0] * inv; lo4.y = acc[n][1] * inv;
        lo4.z = acc[n][2] * inv; lo4.w = acc[n][3] * inv;
        hi4.x = acc[n][4] * inv; hi4.y = acc[n][5] * inv;
        hi4.z = acc[n][6] * inv; hi4.w = acc[n][7] * inv;
        *(float4*)(orow + n * 16)     = lo4;
        *(float4*)(orow + n * 16 + 4) = hi4;
    }
}

// ---------------------------------------------------------------------------
// Fallback kernel (self-contained f32 loads + inline conversion) — used only
// if the workspace is too small for the f16 staging buffers.
// ---------------------------------------------------------------------------
__global__ __launch_bounds__(256)
void fa_fwd_causal_kernel(const float* __restrict__ Q, const float* __restrict__ K,
                          const float* __restrict__ V, float* __restrict__ O)
{
    __shared__ __align__(16) _Float16 plds[WAVES_PER_BLOCK][QTILE][LDS_STRIDE];

    const int lane = threadIdx.x & 31;
    const int w    = threadIdx.x >> 5;
    const int wid  = blockIdx.x * WAVES_PER_BLOCK + w;
    const int bh   = wid >> 7;
    const int q0   = (wid & 127) << 4;
    const int col  = lane & 15;
    const int hi   = lane >> 4;
    const float scale = 0.125f;

    const float* qbase = Q + ((size_t)bh * SEQ + q0 + col) * HD;
    v16h aQ[2];
#pragma unroll
    for (int c = 0; c < 2; ++c) {
        const int d0 = c * 32 + hi * 16;
#pragma unroll
        for (int i = 0; i < 8; ++i) {
            aQ[c][i]     = (_Float16)(qbase[d0 + i]      * scale);
            aQ[c][i + 8] = (_Float16)(qbase[d0 + 8 + i] * scale);
        }
    }

    v8f acc[4];
#pragma unroll
    for (int n = 0; n < 4; ++n) acc[n] = (v8f){};
    float rowmax[8], rowsum[8];
#pragma unroll
    for (int r = 0; r < 8; ++r) { rowmax[r] = -3.0e38f; rowsum[r] = 0.0f; }

    const float* kbh = K + (size_t)bh * SEQ * HD;
    const float* vbh = V + (size_t)bh * SEQ * HD;
    const int klast = q0 + QTILE - 1;

    for (int kb = 0; kb <= klast; kb += KBLK) {
        v16h bK[2][2];
#pragma unroll
        for (int t = 0; t < 2; ++t) {
            const float* kp_row = kbh + (size_t)(kb + t * 16 + col) * HD + hi * 16;
#pragma unroll
            for (int c = 0; c < 2; ++c) {
                const float4* kp = (const float4*)(kp_row + c * 32);
#pragma unroll
                for (int v4 = 0; v4 < 4; ++v4) {
                    float4 f = kp[v4];
                    bK[c][t][v4 * 4 + 0] = (_Float16)f.x;
                    bK[c][t][v4 * 4 + 1] = (_Float16)f.y;
                    bK[c][t][v4 * 4 + 2] = (_Float16)f.z;
                    bK[c][t][v4 * 4 + 3] = (_Float16)f.w;
                }
            }
        }

        v8f s0 = (v8f){}, s1 = (v8f){};
        s0 = __builtin_amdgcn_wmma_f32_16x16x32_f16(false, aQ[0], false, bK[0][0], (short)0, s0, false, false);
        s0 = __builtin_amdgcn_wmma_f32_16x16x32_f16(false, aQ[1], false, bK[1][0], (short)0, s0, false, false);
        s1 = __builtin_amdgcn_wmma_f32_16x16x32_f16(false, aQ[0], false, bK[0][1], (short)0, s1, false, false);
        s1 = __builtin_amdgcn_wmma_f32_16x16x32_f16(false, aQ[1], false, bK[1][1], (short)0, s1, false, false);

        const bool need_mask = (kb + KBLK - 1 > q0);
#pragma unroll
        for (int r = 0; r < 8; ++r) {
            const int qrow = q0 + r + 8 * hi;
            float v0 = s0[r], v1 = s1[r];
            if (need_mask) {
                if (kb + col      > qrow) v0 = -3.0e38f;
                if (kb + 16 + col > qrow) v1 = -3.0e38f;
            }
            float mt = fmaxf(v0, v1);
            mt = fmaxf(mt, __shfl_xor(mt, 1, 16));
            mt = fmaxf(mt, __shfl_xor(mt, 2, 16));
            mt = fmaxf(mt, __shfl_xor(mt, 4, 16));
            mt = fmaxf(mt, __shfl_xor(mt, 8, 16));
            const float mnew = fmaxf(rowmax[r], mt);
            const float corr = __expf(rowmax[r] - mnew);
            rowmax[r] = mnew;
            const float p0 = __expf(v0 - mnew);
            const float p1 = __expf(v1 - mnew);
            float ps = p0 + p1;
            ps += __shfl_xor(ps, 1, 16);
            ps += __shfl_xor(ps, 2, 16);
            ps += __shfl_xor(ps, 4, 16);
            ps += __shfl_xor(ps, 8, 16);
            rowsum[r] = rowsum[r] * corr + ps;
#pragma unroll
            for (int n = 0; n < 4; ++n) acc[n][r] *= corr;
            plds[w][r + 8 * hi][col]      = (_Float16)p0;
            plds[w][r + 8 * hi][col + 16] = (_Float16)p1;
        }

        asm volatile("s_wait_dscnt 0" ::: "memory");

        v16h aP;
        {
            const _Float16* prow = &plds[w][col][hi * 16];
            const v8h lo  = *(const v8h*)(prow);
            const v8h hi8 = *(const v8h*)(prow + 8);
#pragma unroll
            for (int i = 0; i < 8; ++i) { aP[i] = lo[i]; aP[i + 8] = hi8[i]; }
        }

        const float* vp0 = vbh + (size_t)(kb + hi * 16) * HD + col;
#pragma unroll
        for (int n = 0; n < 4; ++n) {
            v16h bv;
            const float* vp = vp0 + n * 16;
#pragma unroll
            for (int i = 0; i < 16; ++i) bv[i] = (_Float16)vp[(size_t)i * HD];
            acc[n] = __builtin_amdgcn_wmma_f32_16x16x32_f16(false, aP, false, bv, (short)0, acc[n], false, false);
        }
    }

    float* obase = O + ((size_t)bh * SEQ + q0) * HD;
#pragma unroll
    for (int r = 0; r < 8; ++r) {
        const float inv = 1.0f / rowsum[r];
        float* orow = obase + (size_t)(r + 8 * hi) * HD + col;
#pragma unroll
        for (int n = 0; n < 4; ++n) orow[n * 16] = acc[n][r] * inv;
    }
}

// ---------------------------------------------------------------------------
extern "C" void kernel_launch(void* const* d_in, const int* in_sizes, int n_in,
                              void* d_out, int out_size, void* d_ws, size_t ws_size,
                              hipStream_t stream) {
    (void)in_sizes; (void)n_in; (void)out_size;
    const float* Q = (const float*)d_in[0];
    const float* K = (const float*)d_in[1];
    const float* V = (const float*)d_in[2];
    float* Out = (float*)d_out;

    const int total_waves = BH_TOT * (SEQ / QTILE);          // 4096
    const int blocks = total_waves / WAVES_PER_BLOCK;        // 512

    const size_t need = 2 * ELEMS * sizeof(_Float16);        // Kh + Vt = 16.8 MB
    if (ws_size >= need) {
        _Float16* Kh = (_Float16*)d_ws;
        _Float16* Vt = Kh + ELEMS;
        prep_kh_kernel<<<(int)(ELEMS / (256 * 8)), 256, 0, stream>>>(K, Kh);
        prep_vt_kernel<<<BH_TOT * (SEQ / 64), 256, 0, stream>>>(V, Vt);
        fa_fwd_causal_ws_kernel<<<blocks, WAVES_PER_BLOCK * 32, 0, stream>>>(Q, Kh, Vt, Out);
    } else {
        fa_fwd_causal_kernel<<<blocks, WAVES_PER_BLOCK * 32, 0, stream>>>(Q, K, V, Out);
    }
}
